// SLULatticeRNN_28132035788899
// MI455X (gfx1250) — matrix operand order repaired and checked by reading, not compiled, not run on hardware
//
#include <hip/hip_runtime.h>
#include <hip/hip_bf16.h>
#include <math.h>

typedef _Float16 f16;
typedef __attribute__((ext_vector_type(16))) _Float16 v16h;
typedef __attribute__((ext_vector_type(8)))  _Float16 v8h;
typedef __attribute__((ext_vector_type(8)))  float    v8f;

#define B_   32
#define T_   128
#define K_   4
#define E_   300
#define EP_  320           // E padded to multiple of 32 for WMMA K
#define H_   512
#define G4_  2048          // 4*H
#define L_   64
#define TB_  (T_*B_)       // 4096
#define NWG_ 16            // workgroups cooperating per recurrence layer
#define SL_  (H_/NWG_)     // h-slice width per WGP = 32
#define NC_  (4*SL_)       // local gate columns per WGP = 128

__device__ __forceinline__ float sigmoidf_(float x){ return 1.0f/(1.0f + __expf(-x)); }

// ---------------- WMMA fragment loaders (contiguous k order: k = 16*(lane/16) + s, same for A and B) ----------------
// A: 16x32 f16, row-major source with leading dim ld (elements).
// lane<16: row=lane,    K = 0..15
// lane>=16: row=lane-16, K = 16..31
__device__ __forceinline__ v16h load_a_frag(const f16* A, int row0, int k0, int ld){
  int lane = threadIdx.x & 31;
  const f16* p = A + (size_t)(row0 + (lane & 15)) * ld + k0 + ((lane >> 4) << 4);
  return *(const v16h*)p;
}

// B: 32x16 f16 from Bt row-major [N, ld] (Bt[n][k] == B[k][n]).
// lane holds column n0+(lane&15); K = (lane>>4)*16 .. +15 contiguous.
__device__ __forceinline__ v16h load_b_frag(const f16* Bt, int n0, int k0, int ld){
  int lane = threadIdx.x & 31;
  return *(const v16h*)(Bt + (size_t)(n0 + (lane & 15)) * ld + k0 + ((lane >> 4) << 4));
}

// ---------------- prep kernels ----------------
__global__ void zero_u32_kernel(unsigned* __restrict__ p, int n){
  int i = blockIdx.x * blockDim.x + threadIdx.x;
  if (i < n) p[i] = 0u;
}

__global__ void pad_cvt_f16_kernel(const float* __restrict__ src, f16* __restrict__ dst,
                                   int rows, int kin, int kout){
  int i = blockIdx.x * blockDim.x + threadIdx.x;
  if (i >= rows * kout) return;
  int r = i / kout, k = i - r * kout;
  dst[i] = (k < kin) ? (f16)src[(size_t)r * kin + k] : (f16)0.0f;
}

__global__ void embed_gather_kernel(const int* __restrict__ inputs,
                                    const float* __restrict__ emb,
                                    f16* __restrict__ x0){
  // x0: [T*B, EP], row r = t*B + b  (matches x transposed to [T,B,E])
  int i = blockIdx.x * blockDim.x + threadIdx.x;
  if (i >= TB_ * EP_) return;
  int r = i / EP_, k = i - r * EP_;
  int t = r / B_, b = r - t * B_;
  int id = inputs[b * T_ + t];
  x0[i] = (k < E_) ? (f16)emb[(size_t)id * E_ + k] : (f16)0.0f;
}

// ---------------- batched x-projection GEMM: C[M,N] = A[M,Kp] @ Bt[N,Kp]^T + bias ----
__global__ void __launch_bounds__(256)
gemm_xproj_kernel(const f16* __restrict__ A, const f16* __restrict__ Bt,
                  const float* __restrict__ bias, float* __restrict__ C,
                  int M, int N, int Kp){
  int wave = blockIdx.x * (blockDim.x >> 5) + (threadIdx.x >> 5);
  int lane = threadIdx.x & 31;
  int ncolGroups = N >> 6;                 // N / 64
  int mtile = wave / ncolGroups;
  int n0    = (wave - mtile * ncolGroups) << 6;
  if (mtile >= (M >> 4)) return;           // wave-uniform

  v8f a0 = {}, a1 = {}, a2 = {}, a3 = {};
  for (int k0 = 0; k0 < Kp; k0 += 32){
    v16h af = load_a_frag(A, mtile << 4, k0, Kp);
    v16h b0 = load_b_frag(Bt, n0 +  0, k0, Kp);
    v16h b1 = load_b_frag(Bt, n0 + 16, k0, Kp);
    v16h b2 = load_b_frag(Bt, n0 + 32, k0, Kp);
    v16h b3 = load_b_frag(Bt, n0 + 48, k0, Kp);
    a0 = __builtin_amdgcn_wmma_f32_16x16x32_f16(false, af, false, b0, (short)0, a0, false, false);
    a1 = __builtin_amdgcn_wmma_f32_16x16x32_f16(false, af, false, b1, (short)0, a1, false, false);
    a2 = __builtin_amdgcn_wmma_f32_16x16x32_f16(false, af, false, b2, (short)0, a2, false, false);
    a3 = __builtin_amdgcn_wmma_f32_16x16x32_f16(false, af, false, b3, (short)0, a3, false, false);
  }
  int hi = lane >> 4, nc = lane & 15;
#pragma unroll
  for (int i = 0; i < 8; ++i){
    int m = (mtile << 4) + (hi << 3) + i;
    float* crow = C + (size_t)m * N;
    crow[n0 +  0 + nc] = a0[i] + bias[n0 +  0 + nc];
    crow[n0 + 16 + nc] = a1[i] + bias[n0 + 16 + nc];
    crow[n0 + 32 + nc] = a2[i] + bias[n0 + 32 + nc];
    crow[n0 + 48 + nc] = a3[i] + bias[n0 + 48 + nc];
  }
}

// ---------------- multi-WGP lattice-LSTM recurrence (one layer) ----------------
// 16 cooperating workgroups; WG p owns h-slice [p*32, p*32+32) of all 4 gates.
// LDS per WG: hprev f16 [32][512] (32KB) + Whh slice f16 [128][512] (128KB)
//           + gate acc f32 [32][128] (16KB) = 180224 B
__global__ void __launch_bounds__(512)
lattice_recur_kernel(const float* __restrict__ gates_x,   // [T*B, 4H], bias folded in
                     const f16*   __restrict__ WhhT,      // [4H, H] row-major f16
                     const int*   __restrict__ prev_idx,  // [B,T,K]
                     const float* __restrict__ prev_marg, // [B,T,K]
                     float* __restrict__ hs,              // [T,B,H]
                     float* __restrict__ cs,              // [T,B,H]
                     f16*   __restrict__ hf16,            // [T*B, H]
                     unsigned* __restrict__ bar){         // arrive counter (zeroed)
  extern __shared__ char smem[];
  f16*   hprev   = (f16*)smem;                                  // [32][512]
  f16*   whh_lds = (f16*)(smem + B_ * H_ * 2);                  // [128][512]
  float* gacc    = (float*)(smem + B_ * H_ * 2 + NC_ * H_ * 2); // [32][128]

  const int tid  = threadIdx.x;
  const int lane = tid & 31;
  const int wave = tid >> 5;            // 0..15
  const int p    = blockIdx.x;          // 0..15 (WGP slice)
  const int b    = tid >> 4;            // 0..31 (mix phase)
  const int hb   = (tid & 15) << 5;     // 32 h's per thread (mix phase)

  const int mtile = wave >> 3;          // 0..1
  const int nl0   = (wave & 7) << 4;    // local gate-column tile base, 0..112

  // ---- preload this WGP's Whh slice into LDS (once) ----
  // local row r in [0,128): global row = (r>>5)*512 + p*32 + (r&31)
  for (int i = tid; i < (NC_ * H_) / 8; i += 512){
    int idx = i << 3;                   // half index into slice
    int r = idx >> 9;                   // /512
    int k = idx & (H_ - 1);
    int grow = ((r >> 5) << 9) + (p << 5) + (r & 31);
    *(v8h*)(whh_lds + idx) = *(const v8h*)(WhhT + (size_t)grow * H_ + k);
  }
  __syncthreads();

  for (int t = 0; t < T_; ++t){
    // ---- phase A: full h_prev = sum_k marg*hs[idx] -> LDS (f16) ----
    {
      int base = (b * T_ + t) * K_;
      int i0 = prev_idx[base+0], i1 = prev_idx[base+1];
      int i2 = prev_idx[base+2], i3 = prev_idx[base+3];
      float m0 = prev_marg[base+0], m1 = prev_marg[base+1];
      float m2 = prev_marg[base+2], m3 = prev_marg[base+3];
      const float* r0 = hs + ((size_t)i0 * B_ + b) * H_ + hb;
      const float* r1 = hs + ((size_t)i1 * B_ + b) * H_ + hb;
      const float* r2 = hs + ((size_t)i2 * B_ + b) * H_ + hb;
      const float* r3 = hs + ((size_t)i3 * B_ + b) * H_ + hb;
#pragma unroll 8
      for (int j = 0; j < 32; ++j){
        float hp = m0*r0[j] + m1*r1[j] + m2*r2[j] + m3*r3[j];
        hprev[b * H_ + hb + j] = (f16)hp;
      }
    }
    __syncthreads();

    // ---- phase B: local gates = h_prev @ Whh_slice^T (all operands in LDS) ----
    {
      v8f acc = {};
      for (int k0 = 0; k0 < H_; k0 += 32){
        v16h af = load_a_frag(hprev, mtile << 4, k0, H_);
        v16h bf = load_b_frag(whh_lds, nl0, k0, H_);
        acc = __builtin_amdgcn_wmma_f32_16x16x32_f16(false, af, false, bf, (short)0, acc, false, false);
      }
      int hi = lane >> 4, nc = lane & 15;
#pragma unroll
      for (int i = 0; i < 8; ++i){
        int m = (mtile << 4) + (hi << 3) + i;
        gacc[m * NC_ + nl0 + nc] = acc[i];
      }
    }
    __syncthreads();

    // ---- phase C: LSTM cell for this WGP's h-slice ----
#pragma unroll
    for (int r = 0; r < (B_ * SL_) / 512; ++r){   // 2 elements per thread
      int o  = tid * 2 + r;
      int bb = o >> 5;                 // batch 0..31
      int hl = o & (SL_ - 1);          // 0..31 within slice
      int h  = p * SL_ + hl;           // global h
      int base = (bb * T_ + t) * K_;
      int i0 = prev_idx[base+0], i1 = prev_idx[base+1];
      int i2 = prev_idx[base+2], i3 = prev_idx[base+3];
      float m0 = prev_marg[base+0], m1 = prev_marg[base+1];
      float m2 = prev_marg[base+2], m3 = prev_marg[base+3];
      float cp = m0 * cs[((size_t)i0 * B_ + bb) * H_ + h]
               + m1 * cs[((size_t)i1 * B_ + bb) * H_ + h]
               + m2 * cs[((size_t)i2 * B_ + bb) * H_ + h]
               + m3 * cs[((size_t)i3 * B_ + bb) * H_ + h];
      const float* gx = gates_x + ((size_t)t * B_ + bb) * G4_;
      const float* gh = gacc + bb * NC_;
      float gi = gh[0*SL_ + hl] + gx[0*H_ + h];
      float gf = gh[1*SL_ + hl] + gx[1*H_ + h];
      float gg = gh[2*SL_ + hl] + gx[2*H_ + h];
      float go = gh[3*SL_ + hl] + gx[3*H_ + h];
      float c  = sigmoidf_(gf) * cp + sigmoidf_(gi) * tanhf(gg);
      float hv = sigmoidf_(go) * tanhf(c);
      cs[((size_t)t * B_ + bb) * H_ + h] = c;
      hs[((size_t)t * B_ + bb) * H_ + h] = hv;
      hf16[((size_t)t * B_ + bb) * H_ + h] = (f16)hv;
    }

    // ---- inter-WGP step barrier (monotonic arrive + acquire spin) ----
    __threadfence();
    __syncthreads();
    if (tid == 0){
      __hip_atomic_fetch_add(bar, 1u, __ATOMIC_RELEASE, __HIP_MEMORY_SCOPE_AGENT);
      unsigned target = (unsigned)(t + 1) * NWG_;
      while (__hip_atomic_load(bar, __ATOMIC_ACQUIRE, __HIP_MEMORY_SCOPE_AGENT) < target){
        __builtin_amdgcn_s_sleep(2);
      }
    }
    __syncthreads();
  }
}

// ---------------- masked max-pool + final linear ----------------
__global__ void __launch_bounds__(1024)
pool_linear_kernel(const float* __restrict__ h2,        // [T,B,H]
                   const int* __restrict__ positions,   // [B,2]
                   const float* __restrict__ lin_w,     // [L,H]
                   const float* __restrict__ lin_b,     // [L]
                   float* __restrict__ out){            // [B,L]
  __shared__ float pooled[B_ * H_];                     // 64KB
  int tid = threadIdx.x;
  {
    int b  = tid >> 5;
    int hb = (tid & 31) << 4;
    int s = positions[b * 2 + 0], e = positions[b * 2 + 1];
#pragma unroll
    for (int j = 0; j < 16; ++j){
      float mx;
      if (s >= e) mx = 0.0f;
      else {
        mx = -3.402823466e+38f;
        for (int t = s; t < e; ++t)
          mx = fmaxf(mx, h2[((size_t)t * B_ + b) * H_ + hb + j]);
      }
      pooled[b * H_ + hb + j] = mx;
    }
  }
  __syncthreads();
#pragma unroll
  for (int r = 0; r < 2; ++r){
    int o = tid * 2 + r;
    int b = o >> 6, l = o & 63;
    const float* pp = pooled + b * H_;
    const float* w  = lin_w + (size_t)l * H_;
    float acc = 0.f;
    for (int k = 0; k < H_; ++k) acc += pp[k] * w[k];
    out[o] = acc + lin_b[l];
  }
}

extern "C" void kernel_launch(void* const* d_in, const int* in_sizes, int n_in,
                              void* d_out, int out_size, void* d_ws, size_t ws_size,
                              hipStream_t stream){
  const int*   inputs    = (const int*)  d_in[0];
  const int*   positions = (const int*)  d_in[1];
  const int*   prev_idx  = (const int*)  d_in[2];
  const float* prev_marg = (const float*)d_in[3];
  const float* emb       = (const float*)d_in[4];
  const float* Wih0      = (const float*)d_in[5];
  const float* Whh0      = (const float*)d_in[6];
  const float* b0        = (const float*)d_in[7];
  const float* Wih1      = (const float*)d_in[8];
  const float* Whh1      = (const float*)d_in[9];
  const float* b1        = (const float*)d_in[10];
  const float* lin_w     = (const float*)d_in[11];
  const float* lin_b     = (const float*)d_in[12];
  float* out = (float*)d_out;

  char* ws = (char*)d_ws;
  size_t off = 0;
  auto alloc = [&](size_t bytes) -> char* {
    char* p = ws + off; off += (bytes + 255) & ~(size_t)255; return p;
  };
  f16*      x0f   = (f16*)     alloc((size_t)TB_ * EP_ * 2);
  f16*      Wih0f = (f16*)     alloc((size_t)G4_ * EP_ * 2);
  f16*      Whh0f = (f16*)     alloc((size_t)G4_ * H_ * 2);
  f16*      Wih1f = (f16*)     alloc((size_t)G4_ * H_ * 2);
  f16*      Whh1f = (f16*)     alloc((size_t)G4_ * H_ * 2);
  float*    gx0   = (float*)   alloc((size_t)TB_ * G4_ * 4);
  float*    gx1   = (float*)   alloc((size_t)TB_ * G4_ * 4);
  float*    hs0   = (float*)   alloc((size_t)TB_ * H_ * 4);
  float*    cs0   = (float*)   alloc((size_t)TB_ * H_ * 4);
  float*    hs1   = (float*)   alloc((size_t)TB_ * H_ * 4);
  float*    cs1   = (float*)   alloc((size_t)TB_ * H_ * 4);
  f16*      h1f   = (f16*)     alloc((size_t)TB_ * H_ * 2);
  f16*      h2f   = (f16*)     alloc((size_t)TB_ * H_ * 2);
  unsigned* bars  = (unsigned*)alloc(2 * sizeof(unsigned));

  const int thr = 256;
  zero_u32_kernel<<<1, 32, 0, stream>>>(bars, 2);
  pad_cvt_f16_kernel<<<(G4_*EP_ + thr-1)/thr, thr, 0, stream>>>(Wih0, Wih0f, G4_, E_, EP_);
  pad_cvt_f16_kernel<<<(G4_*H_  + thr-1)/thr, thr, 0, stream>>>(Whh0, Whh0f, G4_, H_, H_);
  pad_cvt_f16_kernel<<<(G4_*H_  + thr-1)/thr, thr, 0, stream>>>(Wih1, Wih1f, G4_, H_, H_);
  pad_cvt_f16_kernel<<<(G4_*H_  + thr-1)/thr, thr, 0, stream>>>(Whh1, Whh1f, G4_, H_, H_);
  embed_gather_kernel<<<(TB_*EP_ + thr-1)/thr, thr, 0, stream>>>(inputs, emb, x0f);

  // x projections: waves = (M/16)*(N/64) = 256*32 = 8192; 8 waves/block
  const int nblocks = ((TB_/16) * (G4_/64)) / 8;
  gemm_xproj_kernel<<<nblocks, 256, 0, stream>>>(x0f, Wih0f, b0, gx0, TB_, G4_, EP_);

  const size_t lds_bytes = (size_t)B_ * H_ * 2     // hprev
                         + (size_t)NC_ * H_ * 2    // Whh slice
                         + (size_t)B_ * NC_ * 4;   // gate acc  = 180224
  hipFuncSetAttribute(reinterpret_cast<const void*>(lattice_recur_kernel),
                      hipFuncAttributeMaxDynamicSharedMemorySize, (int)lds_bytes);
  lattice_recur_kernel<<<NWG_, 512, lds_bytes, stream>>>(gx0, Whh0f, prev_idx, prev_marg,
                                                         hs0, cs0, h1f, bars + 0);

  gemm_xproj_kernel<<<nblocks, 256, 0, stream>>>(h1f, Wih1f, b1, gx1, TB_, G4_, H_);
  lattice_recur_kernel<<<NWG_, 512, lds_bytes, stream>>>(gx1, Whh1f, prev_idx, prev_marg,
                                                         hs1, cs1, h2f, bars + 1);

  pool_linear_kernel<<<1, 1024, 0, stream>>>(hs1, positions, lin_w, lin_b, out);
}
